// LSTM_15994458211407
// MI455X (gfx1250) — matrix operand order repaired and checked by reading, not compiled, not run on hardware
//
#include <hip/hip_runtime.h>
#include <cstdint>

// Problem constants (fixed by the reference).
#define TT 1024
#define HH 64

typedef __attribute__((ext_vector_type(16))) _Float16 v16h;
typedef __attribute__((ext_vector_type(8)))  _Float16 v8h;
typedef __attribute__((ext_vector_type(8)))  float    v8f;

__device__ __forceinline__ float fast_rcp(float x) { return __builtin_amdgcn_rcpf(x); }

__device__ __forceinline__ float sigm(float x) {
  return fast_rcp(1.0f + __expf(-x));           // v_exp + v_rcp, no precise-div chain
}
// Overflow-safe tanh via exp(-2|x|), fast rcp.
__device__ __forceinline__ float tanh_fast(float x) {
  float a = fabsf(x);
  float e = __expf(-2.0f * a);
  float t = (1.0f - e) * fast_rcp(1.0f + e);
  return copysignf(t, x);
}

// Identity storage: column j stored at j, so each lane's WMMA A-fragment
// (lanes<16 K=0..15, lanes>=16 K=16..31 of the chunk, matching B) is one
// contiguous 32B LDS read.
__device__ __forceinline__ int permcol(int j) {
  return j;
}

// ---- WMMA fragment builders (wave32) ---------------------------------------
// B fragment 32x16 f16 (one 32-K chunk): lane holds output column n = lane%16;
// lanes 0-15 carry K = kbase+0..15 (VGPR v <- K=2v,2v+1), lanes 16-31 carry +16.
__device__ __forceinline__ v16h bfrag_f32row(const float* __restrict__ p) {
  v16h r;
#pragma unroll
  for (int i = 0; i < 16; ++i) r[i] = (_Float16)p[i];
  return r;
}

// A fragment 16x32 f16 from a row-major LDS tile hb[16][64]: single 32B read.
__device__ __forceinline__ v16h afrag_lds(const _Float16* __restrict__ hb,
                                          int m, int hi, int chunk) {
  return *(const v16h*)(hb + m * HH + chunk * 32 + hi * 16);
}

// ============================================================================
// Layer 0: x[B,T,3] -> h1 (f16, row-major tile layout [bt][t][16][64]) to ws.
// 16 WGs (one per 16 batch rows) x 4 waves; wave w owns hidden slice 16w..16w+15
// and computes all four gate tiles for it. One barrier per timestep.
// ============================================================================
__global__ __launch_bounds__(128, 1)
void lstm_l0_kernel(const float* __restrict__ x,
                    const float* __restrict__ Wih,   // [256,3]
                    const float* __restrict__ Whh,   // [256,64]
                    const float* __restrict__ bih,
                    const float* __restrict__ bhh,
                    _Float16* __restrict__ h1g) {
  const int bt = blockIdx.x, tid = threadIdx.x;
  const int lane = tid & 31, w = tid >> 5;
  const int hi = lane >> 4, col = lane & 15;

  __shared__ __align__(32) _Float16 hbuf[2][16 * HH]; // h_t, double buffered
  __shared__ __align__(16) float    xls[2][16 * 4];   // x_t rows (padded to 4)

  // W_hh B-fragments resident in VGPRs (4 gates x 2 K-chunks x 8 VGPRs = 64 VGPRs)
  v16h whhf[4][2];
  float bias[4], wx[4][3];
#pragma unroll
  for (int g = 0; g < 4; ++g) {
    int n = g * 64 + w * 16 + col;   // PyTorch gate order i,f,g,o
#pragma unroll
    for (int c = 0; c < 2; ++c)
      whhf[g][c] = bfrag_f32row(Whh + n * HH + c * 32 + hi * 16);
    bias[g] = bih[n] + bhh[n];
#pragma unroll
    for (int d = 0; d < 3; ++d) wx[g][d] = Wih[n * 3 + d];
  }
  const int pcol = permcol(w * 16 + col);  // scatter column for h writes (identity)

  for (int i = tid; i < 16 * HH; i += 128) hbuf[0][i] = (_Float16)0.0f;
  if (tid < 16) {
#pragma unroll
    for (int d = 0; d < 3; ++d)
      xls[0][tid * 4 + d] = x[((size_t)(bt * 16 + tid) * TT + 0) * 3 + d];
    xls[0][tid * 4 + 3] = 0.0f;
  }
  v8f cst;
#pragma unroll
  for (int r = 0; r < 8; ++r) cst[r] = 0.0f;
  __syncthreads();

  for (int t = 0; t < TT; ++t) {
    const int rb = t & 1, wb = rb ^ 1;

    // Software-pipelined x prefetch (global latency hidden behind the step).
    float xp0 = 0.f, xp1 = 0.f, xp2 = 0.f;
    if (tid < 16) {
      int tn = (t + 1 < TT) ? t + 1 : TT - 1;
      const float* px = x + ((size_t)(bt * 16 + tid) * TT + tn) * 3;
      xp0 = px[0]; xp1 = px[1]; xp2 = px[2];
    }

    // Per-lane x rows for accumulator init (C-tile rows r+8*hi).
    float4 xr[8];
#pragma unroll
    for (int r = 0; r < 8; ++r)
      xr[r] = *(const float4*)&xls[rb][(r + 8 * hi) * 4];

    v16h a0 = afrag_lds(hbuf[rb], col, hi, 0);
    v16h a1 = afrag_lds(hbuf[rb], col, hi, 1);

    v8f gacc[4];
#pragma unroll
    for (int g = 0; g < 4; ++g) {
      v8f acc;
#pragma unroll
      for (int r = 0; r < 8; ++r)
        acc[r] = bias[g] + xr[r].x * wx[g][0] + xr[r].y * wx[g][1] + xr[r].z * wx[g][2];
      acc = __builtin_amdgcn_wmma_f32_16x16x32_f16(false, a0, false, whhf[g][0],
                                                   (short)0, acc, false, false);
      acc = __builtin_amdgcn_wmma_f32_16x16x32_f16(false, a1, false, whhf[g][1],
                                                   (short)0, acc, false, false);
      gacc[g] = acc;
    }

    // Wave-local cell update: all four gates for (row, j=16w+col) are here.
#pragma unroll
    for (int r = 0; r < 8; ++r) {
      float i_ = sigm(gacc[0][r]);
      float f_ = sigm(gacc[1][r]);
      float g_ = tanh_fast(gacc[2][r]);
      float o_ = sigm(gacc[3][r]);
      float c = f_ * cst[r] + i_ * g_;
      cst[r] = c;
      float h = o_ * tanh_fast(c);
      hbuf[wb][(r + 8 * hi) * HH + pcol] = (_Float16)h;
    }
    if (tid < 16) {
      xls[wb][tid * 4 + 0] = xp0; xls[wb][tid * 4 + 1] = xp1;
      xls[wb][tid * 4 + 2] = xp2;
    }
    __syncthreads();

    // Stream h_t to global f16 (row-major tile layout preserved), 2KB coalesced.
    v8h hvv = ((const v8h*)hbuf[wb])[tid];
    ((v8h*)(h1g + ((size_t)bt * TT + t) * 1024))[tid] = hvv;
  }
}

// ============================================================================
// Layer 1 + fused output projection. Input GEMM fused into the step: per gate
// tile K = 128 = [h_{t-1} ; x_t], 4 WMMAs. x_t tiles are prefetched from the
// layer-0 stream with CDNA5 async global->LDS DMA (ASYNCcnt), double buffered.
// ============================================================================
__global__ __launch_bounds__(128, 1)
void lstm_l1_kernel(const _Float16* __restrict__ h1g, // [16][T][16][64] f16, row-major
                    const float* __restrict__ Wih,    // [256,64]
                    const float* __restrict__ Whh,    // [256,64]
                    const float* __restrict__ bih,
                    const float* __restrict__ bhh,
                    const float* __restrict__ Wout,   // [3,64]
                    const float* __restrict__ bout,   // [3]
                    float* __restrict__ out) {        // [B,T,3]
  const int bt = blockIdx.x, tid = threadIdx.x;
  const int lane = tid & 31, w = tid >> 5;
  const int hi = lane >> 4, col = lane & 15;

  __shared__ __align__(32) _Float16 hbuf[2][16 * HH];
  __shared__ __align__(32) _Float16 xbuf[2][16 * HH];          // async DMA target
  __shared__ __align__(32) _Float16 wihl[4 * 4 * 2 * 32 * 16]; // W_ih frags, 32B/lane
  __shared__ float woutl[3 * HH];
  __shared__ float boutl[4];

  v16h whhf[4][2];
  float bias[4];
#pragma unroll
  for (int g = 0; g < 4; ++g) {
    int n = g * 64 + w * 16 + col;
#pragma unroll
    for (int c = 0; c < 2; ++c) {
      whhf[g][c] = bfrag_f32row(Whh + n * HH + c * 32 + hi * 16);
      // W_ih fragments parked in LDS (keeps VGPR pressure down; 2x b128/step reload)
      v16h wi = bfrag_f32row(Wih + n * HH + c * 32 + hi * 16);
      *(v16h*)&wihl[(size_t)(((w * 4 + g) * 2 + c) * 32 + lane) * 16] = wi;
    }
    bias[g] = bih[n] + bhh[n];
  }
  // W_out stored via permcol (identity), same column order as the h rows.
  for (int i = tid; i < 3 * HH; i += 128) {
    int o = i / HH, j = i % HH;
    woutl[o * HH + permcol(j)] = Wout[i];
  }
  if (tid < 3) boutl[tid] = bout[tid];
  for (int i = tid; i < 16 * HH; i += 128) hbuf[0][i] = (_Float16)0.0f;
  { // x tile for t=0 (plain load)
    v8h v = ((const v8h*)(h1g + (size_t)bt * TT * 1024))[tid];
    ((v8h*)xbuf[0])[tid] = v;
  }
  const int pcol = permcol(w * 16 + col);
  v8f cst;
#pragma unroll
  for (int r = 0; r < 8; ++r) cst[r] = 0.0f;
  __syncthreads();

  for (int t = 0; t < TT; ++t) {
    const int rb = t & 1, wb = rb ^ 1;

    // Async DMA prefetch of next input tile (2 KB) into xbuf[wb]; wave 0 only.
    // Consumers of xbuf[wb] (step t+1) are gated by s_wait_asynccnt + barrier.
    if (w == 0) {
      int tn = (t + 1 < TT) ? t + 1 : TT - 1;
      const _Float16* gp = h1g + ((size_t)bt * TT + tn) * 1024 + lane * 8;
      unsigned la = (unsigned)(uintptr_t)(&xbuf[wb][0]) + (unsigned)lane * 16u;
      asm volatile(
          "global_load_async_to_lds_b128 %0, %1, off\n\t"
          "global_load_async_to_lds_b128 %0, %1, off offset:512\n\t"
          "global_load_async_to_lds_b128 %0, %1, off offset:1024\n\t"
          "global_load_async_to_lds_b128 %0, %1, off offset:1536"
          :: "v"(la), "v"(gp) : "memory");
    }

    v16h ah0 = afrag_lds(hbuf[rb], col, hi, 0);
    v16h ah1 = afrag_lds(hbuf[rb], col, hi, 1);
    v16h ax0 = afrag_lds(xbuf[rb], col, hi, 0);
    v16h ax1 = afrag_lds(xbuf[rb], col, hi, 1);

    v8f gacc[4];
#pragma unroll
    for (int g = 0; g < 4; ++g) {
      v8f acc;
#pragma unroll
      for (int r = 0; r < 8; ++r) acc[r] = bias[g];
      acc = __builtin_amdgcn_wmma_f32_16x16x32_f16(false, ah0, false, whhf[g][0],
                                                   (short)0, acc, false, false);
      acc = __builtin_amdgcn_wmma_f32_16x16x32_f16(false, ah1, false, whhf[g][1],
                                                   (short)0, acc, false, false);
      v16h wi0 = *(const v16h*)&wihl[(size_t)(((w * 4 + g) * 2 + 0) * 32 + lane) * 16];
      v16h wi1 = *(const v16h*)&wihl[(size_t)(((w * 4 + g) * 2 + 1) * 32 + lane) * 16];
      acc = __builtin_amdgcn_wmma_f32_16x16x32_f16(false, ax0, false, wi0,
                                                   (short)0, acc, false, false);
      acc = __builtin_amdgcn_wmma_f32_16x16x32_f16(false, ax1, false, wi1,
                                                   (short)0, acc, false, false);
      gacc[g] = acc;
    }

#pragma unroll
    for (int r = 0; r < 8; ++r) {
      float i_ = sigm(gacc[0][r]);
      float f_ = sigm(gacc[1][r]);
      float g_ = tanh_fast(gacc[2][r]);
      float o_ = sigm(gacc[3][r]);
      float c = f_ * cst[r] + i_ * g_;
      cst[r] = c;
      float h = o_ * tanh_fast(c);
      hbuf[wb][(r + 8 * hi) * HH + pcol] = (_Float16)h;
    }

    if (w == 0) asm volatile("s_wait_asynccnt 0x0" ::: "memory");
    __syncthreads();

    // Fused output projection: out[b,t,o] = h2_t . W_out[o,:] + b_out[o]
    // (both operands in the same natural K order)
    if (tid < 48) {
      int row = tid / 3, o = tid % 3;
      float s = boutl[o];
      const v8h* hr8 = (const v8h*)&hbuf[wb][row * HH];
      const float* wo = &woutl[o * HH];
#pragma unroll
      for (int jj = 0; jj < 8; ++jj) {
        v8h hv = hr8[jj];
#pragma unroll
        for (int k = 0; k < 8; ++k) s += (float)hv[k] * wo[jj * 8 + k];
      }
      out[((size_t)(bt * 16 + row) * TT + t) * 3 + o] = s;
    }
  }
}

// ============================================================================
extern "C" void kernel_launch(void* const* d_in, const int* in_sizes, int n_in,
                              void* d_out, int out_size, void* d_ws, size_t ws_size,
                              hipStream_t stream) {
  const float* x    = (const float*)d_in[0];
  const float* Wih0 = (const float*)d_in[1];
  const float* Whh0 = (const float*)d_in[2];
  const float* bih0 = (const float*)d_in[3];
  const float* bhh0 = (const float*)d_in[4];
  const float* Wih1 = (const float*)d_in[5];
  const float* Whh1 = (const float*)d_in[6];
  const float* bih1 = (const float*)d_in[7];
  const float* bhh1 = (const float*)d_in[8];
  const float* Wout = (const float*)d_in[9];
  const float* bout = (const float*)d_in[10];
  float* out = (float*)d_out;

  // Workspace: h1 stream, f16, row-major tile layout [16][1024][16][64] = 32 MB.
  _Float16* h1g = (_Float16*)d_ws;

  lstm_l0_kernel<<<16, 128, 0, stream>>>(x, Wih0, Whh0, bih0, bhh0, h1g);
  lstm_l1_kernel<<<16, 128, 0, stream>>>(h1g, Wih1, Whh1, bih1, bhh1, Wout, bout, out);
}
